// Sub_Graph_18562848654144
// MI455X (gfx1250) — matrix operand order repaired and checked by reading, not compiled, not run on hardware
//
#include <hip/hip_runtime.h>
#include <hip/hip_bf16.h>

typedef __attribute__((ext_vector_type(16))) __bf16 v16bf;
typedef __attribute__((ext_vector_type(8)))  __bf16 v8bf;
typedef __attribute__((ext_vector_type(8)))  float  v8f;

#define LP 64
#define HP 128
#define LN_EPS 1e-5f

union V16U { v16bf v; v8bf h[2]; };

// reduce across the 16 lanes of a half-wave (wave32: xor masks 1,2,4,8 stay
// inside lanes [0,16) and [16,32) independently)
__device__ __forceinline__ float halfwave_sum16(float x) {
#pragma unroll
  for (int m = 1; m <= 8; m <<= 1) x += __shfl_xor(x, m, 32);
  return x;
}

__global__ __launch_bounds__(128)
void polyline_subgraph_kernel(const float* __restrict__ X0,
                              const int*   __restrict__ lengths,
                              const float* __restrict__ W1, const float* __restrict__ b1,
                              const float* __restrict__ g1, const float* __restrict__ be1,
                              const float* __restrict__ W2, const float* __restrict__ b2,
                              const float* __restrict__ g2, const float* __restrict__ be2,
                              const float* __restrict__ W3, const float* __restrict__ b3,
                              const float* __restrict__ g3, const float* __restrict__ be3,
                              float* __restrict__ out)
{
  __shared__ __align__(16) __bf16 sX[LP * HP];   // activations, row-major (64x128) = 16 KB
  __shared__ __align__(16) __bf16 sW[HP * HP];   // W^T: sW[n*HP + k]           = 32 KB
  __shared__ float sPhi[HP];                     // masked column max
  __shared__ float sC[HP];                       // per-column bias (+ phi @ W_phi)

  const int p    = blockIdx.x;
  const int tid  = threadIdx.x;        // 0..127
  const int lane = tid & 31;
  const int wid  = tid >> 5;           // 0..3 : M-tile owned by this wave
  const int len  = lengths[p];         // guaranteed >= 1

  // ---------------- stage input activations: f32 global -> bf16 LDS --------
  {
    const float4* src = (const float4*)(X0 + (size_t)p * (LP * HP));
#pragma unroll 4
    for (int i = tid; i < (LP * HP) / 4; i += 128) {
      float4 f = src[i];
      __bf16* d = &sX[i * 4];
      d[0] = (__bf16)f.x; d[1] = (__bf16)f.y; d[2] = (__bf16)f.z; d[3] = (__bf16)f.w;
    }
  }
  __syncthreads();

  // ---------------- fused layer: X@Wx + (b + phi@Wphi) -> LN -> ReLU -------
  auto layer = [&](const float* Wx, const float* Wphi,
                   const float* bb, const float* gg, const float* bbe,
                   bool have_phi) {
    // per-column bias vector: c[n] = b[n] + sum_k phi[k] * Wphi[k][n]
    {
      float c = bb[tid];
      if (have_phi) {
#pragma unroll 4
        for (int k = 0; k < HP; ++k) c += sPhi[k] * Wphi[k * HP + tid];
      }
      sC[tid] = c;
    }
    // stage W^T into LDS as bf16 (coalesced global read, scattered LDS write)
    for (int k = 0; k < HP; ++k)
      sW[tid * HP + k] = (__bf16)Wx[k * HP + tid];
    __syncthreads();

    // per-lane LN scale/shift + bias init for the 8 N-tiles this wave owns
    float gv[8], bev[8], cv[8];
#pragma unroll
    for (int t = 0; t < 8; ++t) {
      const int col = t * 16 + (lane & 15);
      gv[t] = gg[col]; bev[t] = bbe[col]; cv[t] = sC[col];
    }

    v8f acc[8];
#pragma unroll
    for (int t = 0; t < 8; ++t)
#pragma unroll
      for (int j = 0; j < 8; ++j) acc[t][j] = cv[t];   // D = A*B + C, C = bias

    // 16x16x32 bf16 WMMA: M-tile = wid, all 8 N-tiles, K = 128 in 4 chunks
#pragma unroll
    for (int kc = 0; kc < 4; ++kc) {
      // A fragment (16x32): lane<16 -> row=wid*16+lane, K = kc*32 + {0..7,16..23}
      //                     lane>=16 -> same row set,   K = kc*32 + {8..15,24..31}
      V16U a;
      const int arow = wid * 16 + (lane & 15);
      const int akb  = kc * 32 + ((lane >> 4) << 3);
      a.h[0] = *(const v8bf*)&sX[arow * HP + akb];
      a.h[1] = *(const v8bf*)&sX[arow * HP + akb + 16];
#pragma unroll
      for (int t = 0; t < 8; ++t) {
        // B fragment (32x16) from W^T: lane<16 -> col=lane, K = kc*32 + {0..7,16..23}
        //                              lane>=16 -> col=lane-16, K = kc*32 + {8..15,24..31}
        V16U bf;
        const int bcol = t * 16 + (lane & 15);
        const int bkb  = kc * 32 + ((lane >> 4) << 3);
        bf.h[0] = *(const v8bf*)&sW[bcol * HP + bkb];
        bf.h[1] = *(const v8bf*)&sW[bcol * HP + bkb + 16];
        acc[t] = __builtin_amdgcn_wmma_f32_16x16x32_bf16(
            false, a.v, false, bf.v, (short)0, acc[t], false, false);
      }
    }
    __syncthreads();   // all reads of sX done before we overwrite it

    // in-register LayerNorm + ReLU, write bf16 activations back to sX.
    // Accumulator slot j holds row (wid*16 + j) on lanes 0..15 and
    // row (wid*16 + 8 + j) on lanes 16..31; columns = t*16 + (lane&15).
#pragma unroll
    for (int j = 0; j < 8; ++j) {
      float s = 0.f, sq = 0.f;
#pragma unroll
      for (int t = 0; t < 8; ++t) { const float v = acc[t][j]; s += v; sq += v * v; }
      s  = halfwave_sum16(s);
      sq = halfwave_sum16(sq);
      const float mu   = s * (1.f / HP);
      const float var  = sq * (1.f / HP) - mu * mu;
      const float rstd = __frsqrt_rn(var + LN_EPS);
      const int row = wid * 16 + j + ((lane >> 4) << 3);
#pragma unroll
      for (int t = 0; t < 8; ++t) {
        float v = (acc[t][j] - mu) * rstd * gv[t] + bev[t];
        v = fmaxf(v, 0.f);
        sX[row * HP + t * 16 + (lane & 15)] = (__bf16)v;
      }
    }
    __syncthreads();

    // phi = masked column-max over valid rows (post-ReLU, so >= 0)
    {
      float m = -__builtin_inff();
      for (int r = 0; r < len; ++r) m = fmaxf(m, (float)sX[r * HP + tid]);
      sPhi[tid] = m;
    }
    __syncthreads();
  };

  layer(W1, nullptr,      b1, g1, be1, false);
  layer(W2, W2 + HP * HP, b2, g2, be2, true);
  layer(W3, W3 + HP * HP, b3, g3, be3, true);

  // final pooled features: max_rows(cat(x3, phi3)) == cat(phi3, phi3)
  float* o = out + (size_t)p * (2 * HP);
  const float ph = sPhi[tid];
  o[tid]      = ph;
  o[HP + tid] = ph;
}

extern "C" void kernel_launch(void* const* d_in, const int* in_sizes, int n_in,
                              void* d_out, int out_size, void* d_ws, size_t ws_size,
                              hipStream_t stream) {
  const float* X0  = (const float*)d_in[0];
  const int*   len = (const int*)  d_in[1];
  const float* W1  = (const float*)d_in[2];
  const float* b1  = (const float*)d_in[3];
  const float* g1  = (const float*)d_in[4];
  const float* be1 = (const float*)d_in[5];
  const float* W2  = (const float*)d_in[6];
  const float* b2  = (const float*)d_in[7];
  const float* g2  = (const float*)d_in[8];
  const float* be2 = (const float*)d_in[9];
  const float* W3  = (const float*)d_in[10];
  const float* b3  = (const float*)d_in[11];
  const float* g3  = (const float*)d_in[12];
  const float* be3 = (const float*)d_in[13];
  float* out = (float*)d_out;

  const int P = in_sizes[1];   // one block per polyline, 4 waves each
  polyline_subgraph_kernel<<<P, 128, 0, stream>>>(
      X0, len, W1, b1, g1, be1, W2, b2, g2, be2, W3, b3, g3, be3, out);
}
